// Text2Motion_Transformer_54331336294532
// MI455X (gfx1250) — matrix operand order repaired and checked by reading, not compiled, not run on hardware
//
#include <hip/hip_runtime.h>
#include <math.h>

// ---------------------------------------------------------------------------
// Types for CDNA5 WMMA (wave32)
// ---------------------------------------------------------------------------
typedef __bf16 bf16_t;
typedef bf16_t v16bf __attribute__((ext_vector_type(16)));
typedef bf16_t v8bf  __attribute__((ext_vector_type(8)));
typedef float  v8f   __attribute__((ext_vector_type(8)));

#define SHUF16(lo, hi) __builtin_shufflevector((lo), (hi), 0,1,2,3,4,5,6,7,8,9,10,11,12,13,14,15)

// Async global->LDS copy (CDNA5). VDST = LDS byte offset (low 32 bits of the
// generic shared-pointer address per ISA aperture mapping), VADDR = 64-bit
// global address. Tracked by ASYNCcnt.
__device__ __forceinline__ void async_b128(void* lds_ptr, const void* gptr) {
  unsigned loff = (unsigned)(size_t)lds_ptr;
  asm volatile("global_load_async_to_lds_b128 %0, %1, off"
               :: "v"(loff), "v"(gptr) : "memory");
}
__device__ __forceinline__ void wait_async0() {
  asm volatile("s_wait_asynccnt 0x0" ::: "memory");
}

__device__ __forceinline__ float pe_val(int t, int c) {
  // sincos positional encoding, D = 512
  const float k = -9.210340371976184f / 512.0f;       // -ln(10000)/D
  float freq = __expf((float)(c & ~1) * k);
  float a = (float)t * freq;
  return (c & 1) ? __cosf(a) : __sinf(a);
}

__device__ __forceinline__ float gelu_exact(float x) {
  return 0.5f * x * (1.0f + erff(x * 0.7071067811865476f));
}

// ---------------------------------------------------------------------------
// Model constants
// ---------------------------------------------------------------------------
#define DMODEL 512
#define FMLP   2048
#define NHEAD  8
#define DHEAD  64
#define NVOCAB 1024
#define NTOKE  1026        // V + 2
#define BT     8192        // B*T
#define NS     40960       // B*T*(DEPTH+1)

// ---------------------------------------------------------------------------
// fp32 -> bf16 bulk conversion (n4 = n/4 float4 groups)
// ---------------------------------------------------------------------------
__global__ __launch_bounds__(256) void k_f2bf(const float* __restrict__ src,
                                              bf16_t* __restrict__ dst, size_t n4) {
  size_t i = (size_t)blockIdx.x * 256 + threadIdx.x;
  if (i < n4) {
    float4 v = ((const float4*)src)[i];
    bf16_t* d = dst + i * 4;
    d[0] = (bf16_t)v.x; d[1] = (bf16_t)v.y;
    d[2] = (bf16_t)v.z; d[3] = (bf16_t)v.w;
  }
}

// ---------------------------------------------------------------------------
// x = latents + PE   (rows = B*T, row n has t = n % 64)
// ---------------------------------------------------------------------------
__global__ __launch_bounds__(256) void k_add_pe(const float* __restrict__ latents,
                                                float* __restrict__ X) {
  int n = blockIdx.x;
  int t = n & 63;
  size_t base = (size_t)n * DMODEL;
  for (int c = threadIdx.x; c < DMODEL; c += 256)
    X[base + c] = latents[base + c] + pe_val(t, c);
}

// ---------------------------------------------------------------------------
// Token embedding + cumsum + PE.  X row n*5 already holds cond (from GEMM).
// ---------------------------------------------------------------------------
__global__ __launch_bounds__(256) void k_embed(float* __restrict__ X,
                                               const float* __restrict__ tok_emb,
                                               const int* __restrict__ idxs) {
  int n = blockIdx.x;                 // 0..8191
  const int* id = idxs + (size_t)n * 4;
  int i0 = id[0], i1 = id[1], i2 = id[2], i3 = id[3];
  float* base = X + (size_t)n * 5 * DMODEL;
  for (int c = threadIdx.x; c < DMODEL; c += 256) {
    base[c] += pe_val(0, c);
    float acc = 0.f;
    acc += tok_emb[((size_t)0 * NTOKE + i0) * DMODEL + c];
    base[1 * DMODEL + c] = acc + pe_val(1, c);
    acc += tok_emb[((size_t)1 * NTOKE + i1) * DMODEL + c];
    base[2 * DMODEL + c] = acc + pe_val(2, c);
    acc += tok_emb[((size_t)2 * NTOKE + i2) * DMODEL + c];
    base[3 * DMODEL + c] = acc + pe_val(3, c);
    acc += tok_emb[((size_t)3 * NTOKE + i3) * DMODEL + c];
    base[4 * DMODEL + c] = acc + pe_val(4, c);
  }
}

// ---------------------------------------------------------------------------
// LayerNorm: one wave (32 lanes) per row of 512; fp32 in, bf16 out.
// ---------------------------------------------------------------------------
__global__ __launch_bounds__(256) void k_layernorm(const float* __restrict__ x,
                                                   const float* __restrict__ g,
                                                   const float* __restrict__ b,
                                                   bf16_t* __restrict__ out, int M) {
  int row  = blockIdx.x * 8 + (threadIdx.x >> 5);
  int lane = threadIdx.x & 31;
  if (row >= M) return;
  const float* xr = x + (size_t)row * DMODEL;
  float s = 0.f, s2 = 0.f;
  #pragma unroll
  for (int i = 0; i < DMODEL / 32; i++) {
    float v = xr[lane + i * 32];
    s += v; s2 += v * v;
  }
  #pragma unroll
  for (int o = 16; o > 0; o >>= 1) {
    s  += __shfl_xor(s,  o, 32);
    s2 += __shfl_xor(s2, o, 32);
  }
  float mu  = s * (1.f / DMODEL);
  float var = s2 * (1.f / DMODEL) - mu * mu;
  float rs  = rsqrtf(var + 1e-5f);
  bf16_t* orow = out + (size_t)row * DMODEL;
  #pragma unroll
  for (int i = 0; i < DMODEL / 32; i++) {
    int c = lane + i * 32;
    orow[c] = (bf16_t)((xr[c] - mu) * rs * g[c] + b[c]);
  }
}

// ---------------------------------------------------------------------------
// WMMA GEMM:  C[m,n] = sum_k A[m,k] * W[n,k] (+bias) (+gelu | +residual)
// A bf16 row-major (lda), W bf16 row-major [N,K] (ldw), C OutT (ldc).
// Block tile 128x128x32, 8 waves arranged 2(M) x 4(N), 4x2 wmma tiles/wave.
// Tiles move global->LDS with async copies (ASYNCcnt), double-buffered.
// Requires M%128==0, N%128==0, K%32==0.
// EPI: 0 = bias, 1 = bias+gelu, 2 = bias+residual add from R
// ---------------------------------------------------------------------------
template <int EPI, typename OutT>
__global__ __launch_bounds__(256) void k_gemm(const bf16_t* __restrict__ A, int lda,
                                              const bf16_t* __restrict__ W, int ldw,
                                              const float* __restrict__ bias,
                                              OutT* __restrict__ C, int ldc,
                                              const float* __restrict__ R, int ldr,
                                              int M, int N, int K) {
  __shared__ alignas(16) bf16_t sa[2][128 * 32];
  __shared__ alignas(16) bf16_t sb[2][128 * 32];

  int m0 = blockIdx.y * 128;
  int n0 = blockIdx.x * 128;
  int tid = threadIdx.x;
  int w = tid >> 5, lane = tid & 31;
  int wm = w >> 2, wn = w & 3;
  int lhalf = lane >> 4, l16 = lane & 15;

  const v8f vzero = {0.f, 0.f, 0.f, 0.f, 0.f, 0.f, 0.f, 0.f};
  v8f acc[4][2];
  #pragma unroll
  for (int mt = 0; mt < 4; mt++)
    #pragma unroll
    for (int nt = 0; nt < 2; nt++) acc[mt][nt] = vzero;

  // one 128x32 bf16 tile = 512 x 16B chunks; 256 threads -> 2 chunks each side
  auto issue_tile = [&](int kt) {
    int buf = kt & 1;
    int k0 = kt << 5;
    #pragma unroll
    for (int i = 0; i < 2; i++) {
      int c = tid + i * 256;            // 0..511
      int row  = c >> 2;
      int part = (c & 3) << 3;          // k-element offset (0,8,16,24)
      async_b128(&sa[buf][row * 32 + part],
                 &A[(size_t)(m0 + row) * lda + k0 + part]);
      async_b128(&sb[buf][row * 32 + part],
                 &W[(size_t)(n0 + row) * ldw + k0 + part]);
    }
  };

  int nk = K >> 5;
  issue_tile(0);
  for (int kt = 0; kt < nk; kt++) {
    int buf = kt & 1;
    wait_async0();
    __syncthreads();
    if (kt + 1 < nk) issue_tile(kt + 1);     // fills other buffer during compute

    // A fragments: lane<16 -> K {0..7,16..23}; lane>=16 -> K {8..15,24..31}
    v16bf afr[4];
    #pragma unroll
    for (int mt = 0; mt < 4; mt++) {
      int arow = wm * 64 + mt * 16 + l16;
      int klo  = lhalf * 8;
      v8bf lo = *(const v8bf*)&sa[buf][arow * 32 + klo];
      v8bf hi = *(const v8bf*)&sa[buf][arow * 32 + klo + 16];
      afr[mt] = SHUF16(lo, hi);
    }
    #pragma unroll
    for (int nt = 0; nt < 2; nt++) {
      // B fragment: lane n -> column N=n%16, K {0..7,16..23} (lane<16) / {8..15,24..31}
      int bcol = wn * 32 + nt * 16 + l16;
      int bk   = lhalf * 8;
      v8bf lo = *(const v8bf*)&sb[buf][bcol * 32 + bk];
      v8bf hi = *(const v8bf*)&sb[buf][bcol * 32 + bk + 16];
      v16bf bfr = SHUF16(lo, hi);
      #pragma unroll
      for (int mt = 0; mt < 4; mt++)
        acc[mt][nt] = __builtin_amdgcn_wmma_f32_16x16x32_bf16(
            false, afr[mt], false, bfr, (short)0, acc[mt][nt], false, false);
    }
  }

  // Epilogue: VGPR r -> row r + 8*(lane>=16), col = lane%16
  #pragma unroll
  for (int mt = 0; mt < 4; mt++) {
    #pragma unroll
    for (int nt = 0; nt < 2; nt++) {
      int gm0 = m0 + wm * 64 + mt * 16 + lhalf * 8;
      int gn  = n0 + wn * 32 + nt * 16 + l16;
      float bv = bias ? bias[gn] : 0.f;
      #pragma unroll
      for (int r = 0; r < 8; r++) {
        float v = acc[mt][nt][r] + bv;
        if (EPI == 1) v = gelu_exact(v);
        if (EPI == 2) v += R[(size_t)(gm0 + r) * ldr + gn];
        C[(size_t)(gm0 + r) * ldc + gn] = (OutT)v;
      }
    }
  }
}

// ---------------------------------------------------------------------------
// Fused causal attention for one (sequence, head). dh = 64, scale = 1/8.
// Q,K,V,Y bf16 [nseq*TSEQ, 512]; head h occupies columns [64h, 64h+64).
// TSEQ = 64 (temporal, async-staged) or 5 (base/head, padded to 16/32).
// ---------------------------------------------------------------------------
template <int TSEQ>
__global__ __launch_bounds__(256) void k_attn(const bf16_t* __restrict__ Q,
                                              const bf16_t* __restrict__ K,
                                              const bf16_t* __restrict__ V,
                                              bf16_t* __restrict__ Y) {
  constexpr int TP = (TSEQ + 15) & ~15;        // padded queries/keys
  constexpr int KP = (TP < 32) ? 32 : TP;      // padded K-dim for att@V
  constexpr int NT = TP / 16;

  __shared__ alignas(16) bf16_t sq[TP * 64];   // [q][d]
  __shared__ alignas(16) bf16_t sk[TP * 64];   // [k][d]
  __shared__ alignas(16) bf16_t svt[64 * KP];  // [d][k]
  __shared__ alignas(16) bf16_t satt[TP * KP]; // [q][k] bf16
  __shared__ float ss[TP * TP];                // scores fp32

  int head = blockIdx.x;
  int seq  = blockIdx.y;
  size_t rb = (size_t)seq * TSEQ;
  int cb = head * DHEAD;
  int tid = threadIdx.x;

  if (TSEQ == 64) {
    // rows are 128B contiguous -> async copy, 8 chunks/row, 512 chunks/matrix
    #pragma unroll
    for (int i = 0; i < 2; i++) {
      int c = tid + i * 256;
      int row  = c >> 3;
      int part = (c & 7) << 3;
      async_b128(&sq[row * 64 + part], &Q[(rb + row) * DMODEL + cb + part]);
      async_b128(&sk[row * 64 + part], &K[(rb + row) * DMODEL + cb + part]);
    }
  } else {
    for (int idx = tid; idx < TP * 64; idx += 256) {
      int r = idx >> 6, c = idx & 63;
      sq[idx] = (r < TSEQ) ? Q[(rb + r) * DMODEL + cb + c] : (bf16_t)0.f;
      sk[idx] = (r < TSEQ) ? K[(rb + r) * DMODEL + cb + c] : (bf16_t)0.f;
    }
  }
  for (int idx = tid; idx < 64 * KP; idx += 256) {
    int d = idx / KP, kk = idx % KP;
    svt[idx] = (kk < TSEQ) ? V[(rb + kk) * DMODEL + cb + d] : (bf16_t)0.f;
  }
  if (TSEQ == 64) wait_async0();
  __syncthreads();

  int w = tid >> 5, lane = tid & 31;
  int lhalf = lane >> 4, l16 = lane & 15;
  const v8f vzero = {0.f, 0.f, 0.f, 0.f, 0.f, 0.f, 0.f, 0.f};

  // ---- phase 1: scores = Q K^T * 0.125, causal mask ----
  for (int tt = w; tt < NT * NT; tt += 8) {       // wave-uniform loop
    int qi = tt / NT, kj = tt % NT;
    v8f acc = vzero;
    #pragma unroll
    for (int kk = 0; kk < 2; kk++) {              // d dim = 64 -> 2 x 32
      int arow = qi * 16 + l16;
      int klo  = kk * 32 + lhalf * 8;
      v8bf alo = *(const v8bf*)&sq[arow * 64 + klo];
      v8bf ahi = *(const v8bf*)&sq[arow * 64 + klo + 16];
      v16bf af = SHUF16(alo, ahi);
      int bcol = kj * 16 + l16;                   // key index
      int bk   = kk * 32 + lhalf * 8;
      v8bf blo = *(const v8bf*)&sk[bcol * 64 + bk];
      v8bf bhi = *(const v8bf*)&sk[bcol * 64 + bk + 16];
      v16bf bf = SHUF16(blo, bhi);
      acc = __builtin_amdgcn_wmma_f32_16x16x32_bf16(
          false, af, false, bf, (short)0, acc, false, false);
    }
    #pragma unroll
    for (int r = 0; r < 8; r++) {
      int qrow = qi * 16 + lhalf * 8 + r;
      int kcol = kj * 16 + l16;
      float v = acc[r] * 0.125f;
      ss[qrow * TP + kcol] =
          (kcol <= qrow && qrow < TSEQ && kcol < TSEQ) ? v : -1e30f;
    }
  }
  __syncthreads();

  // ---- phase 2: row softmax, write bf16 att (padded with zeros) ----
  if (tid < TP) {
    int q = tid;
    if (q < TSEQ) {
      float mx = -1e30f;
      for (int k2 = 0; k2 < TP; k2++) mx = fmaxf(mx, ss[q * TP + k2]);
      float sum = 0.f;
      for (int k2 = 0; k2 < TP; k2++) sum += __expf(ss[q * TP + k2] - mx);
      float rs = 1.0f / sum;
      for (int k2 = 0; k2 < TP; k2++)
        satt[q * KP + k2] = (bf16_t)(__expf(ss[q * TP + k2] - mx) * rs);
      for (int k2 = TP; k2 < KP; k2++) satt[q * KP + k2] = (bf16_t)0.f;
    } else {
      for (int k2 = 0; k2 < KP; k2++) satt[q * KP + k2] = (bf16_t)0.f;
    }
  }
  __syncthreads();

  // ---- phase 3: out = att @ V ----
  for (int tt = w; tt < NT * 4; tt += 8) {        // wave-uniform loop
    int qi = tt >> 2, dj = tt & 3;
    v8f acc = vzero;
    #pragma unroll
    for (int kk = 0; kk < KP / 32; kk++) {
      int arow = qi * 16 + l16;
      int klo  = kk * 32 + lhalf * 8;
      v8bf alo = *(const v8bf*)&satt[arow * KP + klo];
      v8bf ahi = *(const v8bf*)&satt[arow * KP + klo + 16];
      v16bf af = SHUF16(alo, ahi);
      int bcol = dj * 16 + l16;                   // output d column
      int bk   = kk * 32 + lhalf * 8;
      v8bf blo = *(const v8bf*)&svt[bcol * KP + bk];
      v8bf bhi = *(const v8bf*)&svt[bcol * KP + bk + 16];
      v16bf bf = SHUF16(blo, bhi);
      acc = __builtin_amdgcn_wmma_f32_16x16x32_bf16(
          false, af, false, bf, (short)0, acc, false, false);
    }
    #pragma unroll
    for (int r = 0; r < 8; r++) {
      int qrow = qi * 16 + lhalf * 8 + r;
      if (qrow < TSEQ)
        Y[(rb + qrow) * DMODEL + cb + dj * 16 + l16] = (bf16_t)acc[r];
    }
  }
}

// ---------------------------------------------------------------------------
// Host-side helpers
// ---------------------------------------------------------------------------
struct StackP {   // raw fp32 pointers as delivered
  const float *ln1_g, *ln1_b, *ln2_g, *ln2_b;
  const float *Wq, *bq, *Wk, *bk, *Wv, *bv, *Wp, *bp;
  const float *W1, *b1, *W2, *b2;
};
struct StackB {   // bf16 weights + fp32 norm/bias params
  const float *ln1_g, *ln1_b, *ln2_g, *ln2_b;
  const float *bq, *bk, *bv, *bp, *b1, *b2;
  const bf16_t *Wq, *Wk, *Wv, *Wp, *W1, *W2;
};

static StackP mk_stack(void* const* d, int o) {
  StackP p;
  p.ln1_g = (const float*)d[o + 0];  p.ln1_b = (const float*)d[o + 1];
  p.ln2_g = (const float*)d[o + 2];  p.ln2_b = (const float*)d[o + 3];
  p.Wq = (const float*)d[o + 4];     p.bq = (const float*)d[o + 5];
  p.Wk = (const float*)d[o + 6];     p.bk = (const float*)d[o + 7];
  p.Wv = (const float*)d[o + 8];     p.bv = (const float*)d[o + 9];
  p.Wp = (const float*)d[o + 10];    p.bp = (const float*)d[o + 11];
  p.W1 = (const float*)d[o + 12];    p.b1 = (const float*)d[o + 13];
  p.W2 = (const float*)d[o + 14];    p.b2 = (const float*)d[o + 15];
  return p;
}

static const bf16_t* conv_w(const float* src, size_t n, bf16_t*& cur, hipStream_t s) {
  bf16_t* dst = cur; cur += n;
  size_t n4 = n >> 2;
  k_f2bf<<<dim3((unsigned)((n4 + 255) / 256)), dim3(256), 0, s>>>(src, dst, n4);
  return dst;
}

static StackB conv_stack(const StackP& p, int L, bf16_t*& cur, hipStream_t s) {
  StackB b;
  b.ln1_g = p.ln1_g; b.ln1_b = p.ln1_b; b.ln2_g = p.ln2_g; b.ln2_b = p.ln2_b;
  b.bq = p.bq; b.bk = p.bk; b.bv = p.bv; b.bp = p.bp; b.b1 = p.b1; b.b2 = p.b2;
  size_t DD = (size_t)DMODEL * DMODEL, FD = (size_t)FMLP * DMODEL;
  b.Wq = conv_w(p.Wq, (size_t)L * DD, cur, s);
  b.Wk = conv_w(p.Wk, (size_t)L * DD, cur, s);
  b.Wv = conv_w(p.Wv, (size_t)L * DD, cur, s);
  b.Wp = conv_w(p.Wp, (size_t)L * DD, cur, s);
  b.W1 = conv_w(p.W1, (size_t)L * FD, cur, s);
  b.W2 = conv_w(p.W2, (size_t)L * FD, cur, s);
  return b;
}

static void run_layer(float* X, bf16_t* Hb, bf16_t* Qb, bf16_t* Kb, bf16_t* Vb,
                      bf16_t* Yb, bf16_t* M1b, const StackB& p, int i, int Mrows,
                      int TSEQ, int nseq, hipStream_t s) {
  const int D = DMODEL, F = FMLP;
  size_t DD = (size_t)D * D, FD = (size_t)F * D;
  dim3 blk(256);
  dim3 gLN((Mrows + 7) / 8);
  dim3 gD(D / 128, Mrows / 128);
  dim3 gF(F / 128, Mrows / 128);

  k_layernorm<<<gLN, blk, 0, s>>>(X, p.ln1_g + (size_t)i * D, p.ln1_b + (size_t)i * D, Hb, Mrows);
  k_gemm<0, bf16_t><<<gD, blk, 0, s>>>(Hb, D, p.Wq + i * DD, D, p.bq + (size_t)i * D, Qb, D, nullptr, 0, Mrows, D, D);
  k_gemm<0, bf16_t><<<gD, blk, 0, s>>>(Hb, D, p.Wk + i * DD, D, p.bk + (size_t)i * D, Kb, D, nullptr, 0, Mrows, D, D);
  k_gemm<0, bf16_t><<<gD, blk, 0, s>>>(Hb, D, p.Wv + i * DD, D, p.bv + (size_t)i * D, Vb, D, nullptr, 0, Mrows, D, D);
  if (TSEQ == 64)
    k_attn<64><<<dim3(NHEAD, nseq), blk, 0, s>>>(Qb, Kb, Vb, Yb);
  else
    k_attn<5><<<dim3(NHEAD, nseq), blk, 0, s>>>(Qb, Kb, Vb, Yb);
  k_gemm<2, float><<<gD, blk, 0, s>>>(Yb, D, p.Wp + i * DD, D, p.bp + (size_t)i * D, X, D, X, D, Mrows, D, D);
  k_layernorm<<<gLN, blk, 0, s>>>(X, p.ln2_g + (size_t)i * D, p.ln2_b + (size_t)i * D, Hb, Mrows);
  k_gemm<1, bf16_t><<<gF, blk, 0, s>>>(Hb, D, p.W1 + i * FD, D, p.b1 + (size_t)i * F, M1b, F, nullptr, 0, Mrows, F, D);
  k_gemm<2, float><<<gD, blk, 0, s>>>(M1b, F, p.W2 + i * FD, F, p.b2 + (size_t)i * D, X, D, X, D, Mrows, D, F);
}

// ---------------------------------------------------------------------------
// kernel_launch
// Input order (setup_inputs insertion order; nested stack dicts contribute
// their 16 leaves in literal order):
//   0 latents, 1 idxs, 2 tok_emb, 3 cond_W, 4 cond_b,
//   5..20 temporal, 21..36 base, 37..52 head,
//   53 ln_f_g, 54 ln_f_b, 55 head_W, 56 lin_W, 57 lin_b
// ---------------------------------------------------------------------------
extern "C" void kernel_launch(void* const* d_in, const int* in_sizes, int n_in,
                              void* d_out, int out_size, void* d_ws, size_t ws_size,
                              hipStream_t stream) {
  (void)in_sizes; (void)n_in; (void)out_size; (void)ws_size;

  const float* latents = (const float*)d_in[0];
  const int*   idxs    = (const int*)d_in[1];
  const float* tok_emb = (const float*)d_in[2];
  const float* cond_W  = (const float*)d_in[3];
  const float* cond_b  = (const float*)d_in[4];
  StackP tpf = mk_stack(d_in, 5);
  StackP bpf = mk_stack(d_in, 21);
  StackP hpf = mk_stack(d_in, 37);
  const float* ln_f_g = (const float*)d_in[53];
  const float* ln_f_b = (const float*)d_in[54];
  const float* head_W = (const float*)d_in[55];
  const float* lin_W  = (const float*)d_in[56];
  const float* lin_b  = (const float*)d_in[57];
  float* out = (float*)d_out;

  // ---- workspace carve (byte-based) ----
  const size_t SZ1 = (size_t)BT * DMODEL;   // 8192*512
  const size_t SZ2 = (size_t)NS * DMODEL;   // 40960*512
  char* p = (char*)d_ws;
  float*  X1  = (float*)p;  p += SZ1 * 4;   // feature fp32 (kept live)
  float*  X2  = (float*)p;  p += SZ2 * 4;   // base/head residual fp32
  bf16_t* X1b = (bf16_t*)p; p += SZ1 * 2;   // feature bf16
  bf16_t* Hb  = (bf16_t*)p; p += SZ2 * 2;   // LN output bf16
  bf16_t* U   = (bf16_t*)p; p += 4 * SZ2 * 2; // Qb,Kb,Vb,Yb | M1b (NS x 2048)
  bf16_t* Qb = U, *Kb = U + SZ2, *Vb = U + 2 * SZ2, *Yb = U + 3 * SZ2;
  bf16_t* M1b = U;
  bf16_t* wcur = (bf16_t*)p;                // bf16 weight arena

  // ---- one-time (per launch) weight conversion to bf16 ----
  StackB tp = conv_stack(tpf, 8, wcur, stream);
  StackB bp = conv_stack(bpf, 2, wcur, stream);
  StackB hp = conv_stack(hpf, 2, wcur, stream);
  const bf16_t* cond_Wb = conv_w(cond_W, (size_t)DMODEL * DMODEL, wcur, stream);
  const bf16_t* head_Wb = conv_w(head_W, (size_t)5 * NVOCAB * DMODEL, wcur, stream);
  const bf16_t* lin_Wb  = conv_w(lin_W,  (size_t)DMODEL * DMODEL, wcur, stream);

  dim3 blk(256);

  // temporal stack on 8192 rows, seq len 64
  k_add_pe<<<BT, blk, 0, stream>>>(latents, X1);
  for (int i = 0; i < 8; i++)
    run_layer(X1, Hb, Qb, Kb, Vb, Yb, M1b, tp, i, BT, 64, 128, stream);

  // feature -> bf16 copy (used by cond + final linear)
  k_f2bf<<<dim3((unsigned)((SZ1 / 4 + 255) / 256)), blk, 0, stream>>>(X1, X1b, SZ1 / 4);

  // cond -> row n*5 of X2 (ldc = 5*512)
  k_gemm<0, float><<<dim3(DMODEL / 128, BT / 128), blk, 0, stream>>>(
      X1b, DMODEL, cond_Wb, DMODEL, cond_b, X2, 5 * DMODEL, nullptr, 0, BT, DMODEL, DMODEL);
  k_embed<<<BT, blk, 0, stream>>>(X2, tok_emb, idxs);

  // base + head stacks on 40960 rows, seq len 5
  for (int i = 0; i < 2; i++)
    run_layer(X2, Hb, Qb, Kb, Vb, Yb, M1b, bp, i, NS, 5, BT, stream);
  for (int i = 0; i < 2; i++)
    run_layer(X2, Hb, Qb, Kb, Vb, Yb, M1b, hp, i, NS, 5, BT, stream);

  // final LN + logits (5 GEMMs, no bias) + linear head on feature
  k_layernorm<<<dim3(NS / 8), blk, 0, stream>>>(X2, ln_f_g, ln_f_b, Hb, NS);
  for (int s = 0; s < 5; s++)
    k_gemm<0, float><<<dim3(NVOCAB / 128, BT / 128), blk, 0, stream>>>(
        Hb + (size_t)s * DMODEL, 5 * DMODEL,
        head_Wb + (size_t)s * NVOCAB * DMODEL, DMODEL, nullptr,
        out + (size_t)s * NVOCAB, 5 * NVOCAB, nullptr, 0, BT, NVOCAB, DMODEL);
  k_gemm<0, float><<<dim3(DMODEL / 128, BT / 128), blk, 0, stream>>>(
      X1b, DMODEL, lin_Wb, DMODEL, lin_b, out + (size_t)BT * 5 * NVOCAB, DMODEL,
      nullptr, 0, BT, DMODEL, DMODEL);
}
